// Data2VecVisionSelfAttention_13451837571567
// MI455X (gfx1250) — matrix operand order repaired and checked by reading, not compiled, not run on hardware
//
#include <hip/hip_runtime.h>
#include <hip/hip_bf16.h>

// ---- problem constants ----
#define BATCH   64
#define SEQ     197
#define SPAD    208      // SEQ padded to multiple of 16
#define HID     768
#define NHEADS  12
#define DHEAD   64
#define MTOT    (BATCH*SEQ)          // 12608 = 197 * 64
#define QSZ     ((size_t)BATCH*NHEADS*SPAD*DHEAD)   // bf16 elements per Q/K/V buffer

typedef __attribute__((ext_vector_type(16))) __bf16 v16bf;
typedef __attribute__((ext_vector_type(8)))  float  v8f;
typedef __attribute__((ext_vector_type(4)))  unsigned int u32x4;
typedef __attribute__((ext_vector_type(8)))  int          i32x8;
typedef __attribute__((ext_vector_type(4)))  int          i32x4;

union ABF { v16bf v; uint4 u[2]; };

#ifndef __has_builtin
#define __has_builtin(x) 0
#endif
#if __has_builtin(__builtin_amdgcn_tensor_load_to_lds) && __has_builtin(__builtin_amdgcn_s_wait_tensorcnt)
#define USE_TDM 1
#else
#define USE_TDM 0
#endif

__device__ __forceinline__ unsigned short f2bf(float f) {
    unsigned int u = __float_as_uint(f);
    u += 0x7FFFu + ((u >> 16) & 1u);       // round-to-nearest-even
    return (unsigned short)(u >> 16);
}

// convert float4 -> 4 packed bf16 (two packed-cvt ops) and store 8 bytes to LDS
__device__ __forceinline__ void store_bf16x4(unsigned short* dst, float4 x) {
    __hip_bfloat162 lo = __float22bfloat162_rn(make_float2(x.x, x.y));
    __hip_bfloat162 hi = __float22bfloat162_rn(make_float2(x.z, x.w));
    *(__hip_bfloat162*)(dst)     = lo;
    *(__hip_bfloat162*)(dst + 2) = hi;
}

// ---------------------------------------------------------------------------
// zero the bf16 workspace (so SEQ..SPAD-1 pad rows are zero)
// ---------------------------------------------------------------------------
__global__ void zero_ws_kernel(uint4* __restrict__ p, int n4) {
    int i = blockIdx.x * 256 + threadIdx.x;
    if (i < n4) p[i] = make_uint4(0u, 0u, 0u, 0u);
}

// ---------------------------------------------------------------------------
// QKV projection: y = x @ W^T (+ bias), output bf16 in [B, H, SPAD, DHEAD]
// grid = (197, 12, 3)  block = 128 (4 waves). 64x64 tile per block.
// ---------------------------------------------------------------------------
__global__ __launch_bounds__(128) void qkv_kernel(
    const float* __restrict__ hs,
    const float* __restrict__ Wq, const float* __restrict__ bq,
    const float* __restrict__ Wk,
    const float* __restrict__ Wv, const float* __restrict__ bv,
    unsigned short* __restrict__ Qb,
    unsigned short* __restrict__ Kb,
    unsigned short* __restrict__ Vb)
{
    __shared__ __align__(16) unsigned short Xs[64][32];
    __shared__ __align__(16) unsigned short Ws[64][32];

    const int z = blockIdx.z;
    const float* W    = (z == 0) ? Wq : ((z == 1) ? Wk : Wv);
    const float* bias = (z == 0) ? bq : ((z == 2) ? bv : nullptr);
    unsigned short* Ob = (z == 0) ? Qb : ((z == 1) ? Kb : Vb);

    const int t    = threadIdx.x;
    const int wave = t >> 5;
    const int lane = t & 31;
    const int g    = lane >> 4;      // half-wave
    const int lc   = lane & 15;
    const int m0   = blockIdx.x * 64;
    const int n0   = blockIdx.y * 64;

    v8f acc[4] = {};

    for (int kk = 0; kk < HID; kk += 32) {
        __syncthreads();
        // issue all 8 tile loads first so they overlap, then convert+store
        float4 xv[4], wv4[4];
        #pragma unroll
        for (int i = 0; i < 4; ++i) {
            int idx = t + i * 128;          // 0..511 float4 slots
            int row = idx >> 3;
            int c4  = idx & 7;
            xv[i]  = *(const float4*)(hs + (size_t)(m0 + row) * HID + kk + c4 * 4);
            wv4[i] = *(const float4*)(W  + (size_t)(n0 + row) * HID + kk + c4 * 4);
        }
        #pragma unroll
        for (int i = 0; i < 4; ++i) {
            int idx = t + i * 128;
            int row = idx >> 3;
            int c4  = idx & 7;
            store_bf16x4(&Xs[row][c4 * 4], xv[i]);
            store_bf16x4(&Ws[row][c4 * 4], wv4[i]);
        }
        // prefetch next K-step tiles into cache while we compute
        if (kk + 32 < HID) {
            int row = t >> 3, c4 = t & 7;
            __builtin_prefetch(hs + (size_t)(m0 + row) * HID + kk + 32 + c4 * 4, 0, 1);
            __builtin_prefetch(W  + (size_t)(n0 + row) * HID + kk + 32 + c4 * 4, 0, 1);
        }
        __syncthreads();

        // A fragment: row = wave*16 + lc, k halves at g*8 and 16+g*8
        ABF a;
        const int am  = wave * 16 + lc;
        const int klo = g * 8;
        a.u[0] = *(const uint4*)&Xs[am][klo];
        a.u[1] = *(const uint4*)&Xs[am][16 + klo];

        #pragma unroll
        for (int nt = 0; nt < 4; ++nt) {
            ABF bf;
            const unsigned short* wrow = &Ws[nt * 16 + lc][g * 8];
            bf.u[0] = *(const uint4*)(wrow);
            bf.u[1] = *(const uint4*)(wrow + 16);
            acc[nt] = __builtin_amdgcn_wmma_f32_16x16x32_bf16(
                false, a.v, false, bf.v, (short)0, acc[nt], false, false);
        }
    }

    // epilogue: scatter bf16 into [B, H, SPAD, DHEAD]
    size_t baseR[8];
    #pragma unroll
    for (int r = 0; r < 8; ++r) {
        int gm = m0 + wave * 16 + r + g * 8;
        int b  = gm / SEQ;
        int s  = gm - b * SEQ;
        baseR[r] = (size_t)b * (NHEADS * SPAD * DHEAD) + (size_t)s * DHEAD;
    }
    #pragma unroll
    for (int nt = 0; nt < 4; ++nt) {
        int   n    = n0 + nt * 16 + lc;
        float badd = bias ? bias[n] : 0.0f;
        size_t hoff = (size_t)(n >> 6) * (SPAD * DHEAD) + (n & 63);
        #pragma unroll
        for (int r = 0; r < 8; ++r)
            Ob[baseR[r] + hoff] = f2bf(acc[nt][r] + badd);
    }
}

// ---------------------------------------------------------------------------
// Fused attention: scores = (Q K^T)/8 + relpos_bias ; softmax ; ctx = P V
// grid = B*NHEADS blocks, 256 threads (8 waves). One (b,h) per block.
// ---------------------------------------------------------------------------
__global__ __launch_bounds__(256) void attn_kernel(
    const unsigned short* __restrict__ Qb,
    const unsigned short* __restrict__ Kb,
    const unsigned short* __restrict__ Vb,
    const float* __restrict__ bias_table,
    float* __restrict__ out)
{
    __shared__ __align__(16) unsigned short Ks[SPAD][DHEAD];     // 26 KB
    __shared__ __align__(16) unsigned short Vt[DHEAD][224];      // 28 KB (keys padded to 224)
    __shared__ __align__(16) unsigned short Ps[8][16][32];       //  8 KB per-wave probs relayout

    const int bh  = blockIdx.x;          // = b*NHEADS + h
    const int b   = bh / NHEADS;
    const int h   = bh - b * NHEADS;
    const int t   = threadIdx.x;
    const int wave = t >> 5, lane = t & 31, g = lane >> 4, lc = lane & 15;
    const size_t base = (size_t)bh * SPAD * DHEAD;

    // ---- stage K into LDS (row-major [key][d]) via Tensor Data Mover ----
#if USE_TDM
    if (wave == 0) {
        // one contiguous 208x64 bf16 tile = 13312 elements: describe as a 1-row tile
        unsigned long long gaddr = (unsigned long long)(const void*)(Kb + base);
        unsigned int ldsOff = (unsigned int)(unsigned long long)
            (__attribute__((address_space(3))) unsigned short*)&Ks[0][0];
        u32x4 g0;
        g0.x = 0x1u;                                            // count=1, user descriptor
        g0.y = ldsOff;                                          // lds_addr (bytes)
        g0.z = (unsigned int)gaddr;                             // global_addr[31:0]
        g0.w = (unsigned int)((gaddr >> 32) & 0x01FFFFFFull)    // global_addr[56:32]
             | 0x80000000u;                                     // type = 2 ("image")
        i32x8 g1;
        g1[0] = 0x00010000;                 // workgroup_mask=0, data_size=1 (2 bytes)
        g1[1] = (int)((13312u & 0xFFFFu) << 16);  // tensor_dim0[15:0] -> bits 63:48
        g1[2] = (int)(1u << 16);            // tensor_dim0 hi=0 ; tensor_dim1=1
        g1[3] = (int)(13312u << 16);        // tensor_dim1 hi=0 ; tile_dim0=13312
        g1[4] = 0;                          // tile_dim1=0, tile_dim2=0 (unused)
        g1[5] = 13312;                      // tensor_dim0_stride low
        g1[6] = 0;
        g1[7] = 0;
        i32x4 z4 = {0, 0, 0, 0};
        i32x8 z8 = {0, 0, 0, 0, 0, 0, 0, 0};
        __builtin_amdgcn_tensor_load_to_lds(g0, g1, z4, z4, z8, 0);
        __builtin_amdgcn_s_wait_tensorcnt(0);
    }
#else
    {
        const uint4* src = (const uint4*)(Kb + base);
        uint4* dst = (uint4*)&Ks[0][0];
        for (int i = t; i < SPAD * DHEAD / 8; i += 256) dst[i] = src[i];
    }
#endif
    // zero key-pad columns of Vt (208..223)
    for (int i = t; i < DHEAD * 16; i += 256) {
        int d = i >> 4, s = 208 + (i & 15);
        Vt[d][s] = 0;
    }
    // ---- stage V transposed into LDS: Vt[d][key] ----
    {
        const unsigned int* src = (const unsigned int*)(Vb + base);  // 2 bf16 per read
        for (int i = t; i < SPAD * DHEAD / 2; i += 256) {
            unsigned int pv = src[i];
            int s = i >> 5;               // element = 2*i ; key = element / 64
            int d = (i & 31) * 2;
            Vt[d][s]     = (unsigned short)(pv & 0xFFFFu);
            Vt[d + 1][s] = (unsigned short)(pv >> 16);
        }
    }
    __syncthreads();

    const float* btab = bias_table + h;

    for (int qt = wave; qt < 13; qt += 8) {
        // ---- Q A-fragments (two K=32 chunks covering d=0..63) ----
        ABF aQ[2];
        {
            const unsigned short* qrow = Qb + base + (size_t)(qt * 16 + lc) * DHEAD;
            const int klo = g * 8;
            #pragma unroll
            for (int c = 0; c < 2; ++c) {
                aQ[c].u[0] = *(const uint4*)(qrow + c * 32 + klo);
                aQ[c].u[1] = *(const uint4*)(qrow + c * 32 + 16 + klo);
            }
        }

        // per-row linearized relpos components: idx = Aq - Ak (+364 folded into Aq)
        int q0r[8], Aq[8];
        #pragma unroll
        for (int r = 0; r < 8; ++r) {
            int q0  = qt * 16 + r + g * 8;
            int qm1 = q0 - 1;
            q0r[r] = q0;
            Aq[r]  = qm1 + 13 * (qm1 / 14) + 364;
        }

        // ---- scores: 13 key tiles, keep in registers (C-layout) ----
        v8f sc[14];
        sc[13] = (v8f){};                 // virtual zero tile for key pad 208..223
        #pragma unroll
        for (int kt = 0; kt < 13; ++kt) {
            v8f av = {};
            #pragma unroll
            for (int c = 0; c < 2; ++c) {
                ABF bK;
                const unsigned short* krow = &Ks[kt * 16 + lc][c * 32 + g * 8];
                bK.u[0] = *(const uint4*)(krow);
                bK.u[1] = *(const uint4*)(krow + 16);
                av = __builtin_amdgcn_wmma_f32_16x16x32_bf16(
                    false, aQ[c].v, false, bK.v, (short)0, av, false, false);
            }
            // fused scale + relative position bias + key mask
            const int  k0    = kt * 16 + lc;
            const int  km1   = k0 - 1;
            const int  Ak    = km1 + 13 * (km1 / 14);
            const bool kcls  = (k0 == 0);
            const bool kmask = (k0 >= SEQ);
            #pragma unroll
            for (int r = 0; r < 8; ++r) {
                int idx;
                if (q0r[r] == 0) idx = kcls ? 731 : 729;     // cls-cls / cls-to-token
                else             idx = kcls ? 730 : (Aq[r] - Ak);
                float val = av[r] * 0.125f + btab[idx * NHEADS];
                sc[kt][r] = kmask ? -3.0e38f : val;
            }
        }

        // ---- softmax per query row (half-wave shuffle reductions) ----
        #pragma unroll
        for (int r = 0; r < 8; ++r) {
            float mx = sc[0][r];
            #pragma unroll
            for (int kt = 1; kt < 13; ++kt) mx = fmaxf(mx, sc[kt][r]);
            #pragma unroll
            for (int off = 1; off < 16; off <<= 1)
                mx = fmaxf(mx, __shfl_xor(mx, off, 16));
            float sum = 0.0f;
            #pragma unroll
            for (int kt = 0; kt < 13; ++kt) {
                float e = __expf(sc[kt][r] - mx);
                sc[kt][r] = e;
                sum += e;
            }
            #pragma unroll
            for (int off = 1; off < 16; off <<= 1)
                sum += __shfl_xor(sum, off, 16);
            float inv = 1.0f / sum;
            #pragma unroll
            for (int kt = 0; kt < 13; ++kt) sc[kt][r] *= inv;
        }

        // ---- ctx = P @ V : 7 key chunks of 32, relayout P via LDS ----
        v8f o4[4] = {};
        #pragma unroll
        for (int c = 0; c < 7; ++c) {
            // write this wave's P chunk (C-layout -> row-major 16x32 bf16)
            #pragma unroll
            for (int r = 0; r < 8; ++r) {
                int m = r + g * 8;
                Ps[wave][m][lc]      = f2bf(sc[2 * c][r]);
                Ps[wave][m][16 + lc] = f2bf(sc[2 * c + 1][r]);
            }
            asm volatile("s_wait_dscnt 0x0" ::: "memory");
            ABF aP;
            {
                const int klo = g * 8;
                aP.u[0] = *(const uint4*)&Ps[wave][lc][klo];
                aP.u[1] = *(const uint4*)&Ps[wave][lc][16 + klo];
            }
            #pragma unroll
            for (int nt = 0; nt < 4; ++nt) {
                ABF bV;
                const unsigned short* vrow = &Vt[nt * 16 + lc][c * 32 + g * 8];
                bV.u[0] = *(const uint4*)(vrow);
                bV.u[1] = *(const uint4*)(vrow + 16);
                o4[nt] = __builtin_amdgcn_wmma_f32_16x16x32_bf16(
                    false, aP.v, false, bV.v, (short)0, o4[nt], false, false);
            }
            asm volatile("s_wait_dscnt 0x0" ::: "memory");
        }

        // ---- epilogue: [B, S, HID] fp32 ----
        #pragma unroll
        for (int nt = 0; nt < 4; ++nt) {
            #pragma unroll
            for (int r = 0; r < 8; ++r) {
                int q0 = qt * 16 + r + g * 8;
                if (q0 < SEQ) {
                    out[((size_t)b * SEQ + q0) * HID + h * DHEAD + nt * 16 + lc] = o4[nt][r];
                }
            }
        }
    }
}

// ---------------------------------------------------------------------------
extern "C" void kernel_launch(void* const* d_in, const int* in_sizes, int n_in,
                              void* d_out, int out_size, void* d_ws, size_t ws_size,
                              hipStream_t stream) {
    const float* hs = (const float*)d_in[0];
    const float* Wq = (const float*)d_in[1];
    const float* bq = (const float*)d_in[2];
    const float* Wk = (const float*)d_in[3];
    const float* Wv = (const float*)d_in[4];
    const float* bv = (const float*)d_in[5];
    const float* bt = (const float*)d_in[6];
    // d_in[7], d_in[8] = resolution (224,224) -> window 14x14, baked into index math

    unsigned short* Qb = (unsigned short*)d_ws;
    unsigned short* Kb = Qb + QSZ;
    unsigned short* Vb = Kb + QSZ;

    const int n4 = (int)(3 * QSZ * sizeof(unsigned short) / 16);
    zero_ws_kernel<<<(n4 + 255) / 256, 256, 0, stream>>>((uint4*)d_ws, n4);

    dim3 g1(MTOT / 64, HID / 64, 3);   // (197, 12, 3)
    qkv_kernel<<<g1, 128, 0, stream>>>(hs, Wq, bq, Wk, Wv, bv, Qb, Kb, Vb);

    attn_kernel<<<BATCH * NHEADS, 256, 0, stream>>>(Qb, Kb, Vb, bt, (float*)d_out);
}
